// MultiViewGAT_51539607552338
// MI455X (gfx1250) — matrix operand order repaired and checked by reading, not compiled, not run on hardware
//
#include <hip/hip_runtime.h>
#include <stdint.h>

// ---------------- constants ----------------
constexpr int N_   = 12288;
constexpr int Vv_  = 3;
constexpr int FIN_ = 256;
constexpr int H_   = 256;
constexpr int D_   = 128;
constexpr int E_   = 393216;
constexpr int EN_  = E_ + N_;          // edges + self loops
constexpr float NEG_ = 0.2f;
constexpr float EPS_ = 1e-16f;

// ---------------- types for WMMA ----------------
typedef __attribute__((ext_vector_type(16))) __bf16 bf16x16;
typedef __attribute__((ext_vector_type(8)))  float  f32x8;

union Frag16 { bf16x16 v; uint32_t u[8]; };

// async global->LDS path (gfx1250): guarded so a missing builtin can't break the build
#if defined(__gfx1250__) && __has_builtin(__builtin_amdgcn_global_load_async_to_lds_b128)
#define ASYNC_LDS 1
typedef int v4i_ __attribute__((vector_size(16)));           // matches builtin's V4i
typedef __attribute__((address_space(1))) v4i_* as1_v4ip;    // global
typedef __attribute__((address_space(3))) v4i_* as3_v4ip;    // LDS
#else
#define ASYNC_LDS 0
#endif

__device__ __forceinline__ void wait_async0() {
#if ASYNC_LDS
#if __has_builtin(__builtin_amdgcn_s_wait_asynccnt)
    __builtin_amdgcn_s_wait_asynccnt(0);
#else
    asm volatile("s_wait_asynccnt 0x0" ::: "memory");
#endif
#endif
}

// ---------------- helpers ----------------
__device__ __forceinline__ uint16_t f32_to_bf16(float f) {
    uint32_t u = __float_as_uint(f);
    u = u + 0x7FFFu + ((u >> 16) & 1u);   // round-to-nearest-even
    return (uint16_t)(u >> 16);
}

__device__ __forceinline__ void atomicMaxFloat(float* addr, float val) {
    unsigned int* ua = (unsigned int*)addr;
    unsigned int old = __float_as_uint(*addr);
    while (__uint_as_float(old) < val) {
        unsigned int assumed = old;
        old = atomicCAS(ua, assumed, __float_as_uint(val));
        if (old == assumed) break;
    }
}

// ---------------- elementwise utility kernels ----------------
__global__ void k_fill_f32(float* p, float v, size_t n) {
    size_t i = blockIdx.x * (size_t)blockDim.x + threadIdx.x;
    if (i < n) p[i] = v;
}

__global__ void k_cvt_bf16(const float* __restrict__ in, uint16_t* __restrict__ out, size_t n) {
    size_t i = blockIdx.x * (size_t)blockDim.x + threadIdx.x;
    if (i < n) out[i] = f32_to_bf16(in[i]);
}

// Wt[n,k] = bf16(W[k,n]) ;  W is [K, Nc] row-major f32
__global__ void k_transpose_cvt(const float* __restrict__ W, uint16_t* __restrict__ Wt,
                                int K, int Nc) {
    int i = blockIdx.x * blockDim.x + threadIdx.x;
    if (i >= K * Nc) return;
    int k = i / Nc, n = i % Nc;
    Wt[(size_t)n * K + k] = f32_to_bf16(W[i]);
}

// ---------------- WMMA bf16 GEMM (C = A * Bt^T) ----------------
// A  : [M , K]  bf16 row-major
// Bt : [Nc, K]  bf16 row-major   (contiguous K pairs for both fragments)
// C  : [M , Nc] f32              mode 0: plain store, mode 1: sigmoid + NT store
// BM=BN=128, BK=32, 256 threads (8 wave32), wave w owns rows [16w,16w+16).
// Double-buffered LDS; panels streamed with GLOBAL_LOAD_ASYNC_TO_LDS_B128
// (ASYNCcnt) so the copy of tile k+1 overlaps the 8 WMMAs of tile k.
__global__ __launch_bounds__(256) void k_gemm_nt_bf16(
        const uint16_t* __restrict__ A, const uint16_t* __restrict__ Bt,
        float* __restrict__ C, int M, int Nc, int K, int mode) {
    constexpr int BK = 32;
    __shared__ uint16_t lA[2][128 * BK];
    __shared__ uint16_t lB[2][128 * BK];

    const int tid  = threadIdx.x;
    const int wave = tid >> 5;
    const int lane = tid & 31;
    const int hs   = lane >> 4;     // half-select (wave32 WMMA layout)
    const int l15  = lane & 15;

    const int rowBase = blockIdx.y * 128;
    const int colBase = blockIdx.x * 128;

    // this thread's two 16-byte staging slots (512 uint4 slots per panel)
    const int idx0 = tid * 2;
    const int r0 = idx0 >> 2,        o0 = (idx0 & 3) * 8;        // u16 offset
    const int r1 = (idx0 + 1) >> 2,  o1 = ((idx0 + 1) & 3) * 8;

    auto stage = [&](int buf, int kk) {
#if ASYNC_LDS
        __builtin_amdgcn_global_load_async_to_lds_b128(
            (as1_v4ip)(A + (size_t)(rowBase + r0) * K + kk + o0), (as3_v4ip)&lA[buf][r0 * BK + o0], 0, 0);
        __builtin_amdgcn_global_load_async_to_lds_b128(
            (as1_v4ip)(A + (size_t)(rowBase + r1) * K + kk + o1), (as3_v4ip)&lA[buf][r1 * BK + o1], 0, 0);
        __builtin_amdgcn_global_load_async_to_lds_b128(
            (as1_v4ip)(Bt + (size_t)(colBase + r0) * K + kk + o0), (as3_v4ip)&lB[buf][r0 * BK + o0], 0, 0);
        __builtin_amdgcn_global_load_async_to_lds_b128(
            (as1_v4ip)(Bt + (size_t)(colBase + r1) * K + kk + o1), (as3_v4ip)&lB[buf][r1 * BK + o1], 0, 0);
#else
        *(uint4*)&lA[buf][r0 * BK + o0] = *(const uint4*)(A + (size_t)(rowBase + r0) * K + kk + o0);
        *(uint4*)&lA[buf][r1 * BK + o1] = *(const uint4*)(A + (size_t)(rowBase + r1) * K + kk + o1);
        *(uint4*)&lB[buf][r0 * BK + o0] = *(const uint4*)(Bt + (size_t)(colBase + r0) * K + kk + o0);
        *(uint4*)&lB[buf][r1 * BK + o1] = *(const uint4*)(Bt + (size_t)(colBase + r1) * K + kk + o1);
        if (kk + BK < K)
            __builtin_prefetch(A + (size_t)(rowBase + r0) * K + kk + BK, 0, 3);
#endif
    };

    f32x8 acc[8] = {};              // 8 N-subtiles of 16x16, f32 accum

    const int nk = K / BK;
    stage(0, 0);
    for (int ik = 0; ik < nk; ik++) {
        const int buf = ik & 1;
        wait_async0();               // this wave's panel loads landed in LDS
        __syncthreads();             // everyone's landed; prev compute done
        if (ik + 1 < nk) stage(buf ^ 1, (ik + 1) * BK);  // overlap with WMMAs

        // A fragment: 16x32 bf16. lane<16: K {0..7,16..23}; lane>=16: {8..15,24..31}
        Frag16 af;
        const uint32_t* arow = (const uint32_t*)(&lA[buf][(wave * 16 + l15) * BK]);
        #pragma unroll
        for (int v = 0; v < 4; v++) {
            af.u[v]     = arow[v + 4 * hs];
            af.u[4 + v] = arow[8 + v + 4 * hs];
        }
        // 8 WMMAs across the 128-wide N tile
        #pragma unroll
        for (int t = 0; t < 8; t++) {
            Frag16 bf;
            const uint32_t* brow = (const uint32_t*)(&lB[buf][(t * 16 + l15) * BK]);
            #pragma unroll
            for (int v = 0; v < 8; v++) bf.u[v] = brow[(v & 3) + 4 * hs + 8 * (v >> 2)];
            acc[t] = __builtin_amdgcn_wmma_f32_16x16x32_bf16(
                         false, af.v, false, bf.v, (short)0, acc[t], false, false);
        }
    }

    // epilogue: VGPR r / lane<16 -> (M=r, N=lane); lane>=16 -> (M=r+8, N=lane-16)
    #pragma unroll
    for (int t = 0; t < 8; t++) {
        int n = colBase + t * 16 + l15;
        #pragma unroll
        for (int r = 0; r < 8; r++) {
            int m = rowBase + wave * 16 + r + 8 * hs;
            float x = acc[t][r];
            if (mode == 1) {
                x = 1.0f / (1.0f + __expf(-x));
                // 604MB write-once output (> 192MB L2): stream with NT hint
                __builtin_nontemporal_store(x, &C[(size_t)m * (size_t)Nc + n]);
            } else {
                C[(size_t)m * (size_t)Nc + n] = x;
            }
        }
    }
}

// ---------------- GAT edge pipeline ----------------
// e_s[n] = hp[n,:] . a_src ; e_d[n] = hp[n,:] . a_dst   (wave per node)
__global__ void k_node_dots(const float* __restrict__ hp,
                            const float* __restrict__ a_s, const float* __restrict__ a_d,
                            float* __restrict__ e_s, float* __restrict__ e_d,
                            int N, int C) {
    int n    = (blockIdx.x * blockDim.x + threadIdx.x) >> 5;
    int lane = threadIdx.x & 31;
    if (n >= N) return;
    float s = 0.f, d = 0.f;
    for (int c = lane; c < C; c += 32) {
        float h = hp[(size_t)n * C + c];
        s += h * a_s[c];
        d += h * a_d[c];
    }
    #pragma unroll
    for (int off = 16; off; off >>= 1) {
        s += __shfl_down(s, off, 32);
        d += __shfl_down(d, off, 32);
    }
    if (lane == 0) { e_s[n] = s; e_d[n] = d; }
}

__device__ __forceinline__ void edge_sd(const int* __restrict__ es, const int* __restrict__ ed,
                                        int i, int& s, int& t) {
    if (i < E_) { s = es[i]; t = ed[i]; } else { s = i - E_; t = i - E_; }
}

// e = leaky_relu(e_s[src]+e_d[dst]); segmax[dst] = max(...)
__global__ void k_edge_max(const int* __restrict__ es, const int* __restrict__ ed,
                           const float* __restrict__ e_s, const float* __restrict__ e_d,
                           float* __restrict__ ebuf, float* __restrict__ segmax) {
    int i = blockIdx.x * blockDim.x + threadIdx.x;
    if (i >= EN_) return;
    int s, t; edge_sd(es, ed, i, s, t);
    float e = e_s[s] + e_d[t];
    e = (e > 0.f) ? e : NEG_ * e;
    ebuf[i] = e;
    atomicMaxFloat(&segmax[t], e);
}

// ex = exp(e - m[dst]); segsum[dst] += ex
__global__ void k_edge_expsum(const int* __restrict__ es, const int* __restrict__ ed,
                              float* __restrict__ ebuf,
                              const float* __restrict__ segmax, float* __restrict__ segsum) {
    int i = blockIdx.x * blockDim.x + threadIdx.x;
    if (i >= EN_) return;
    int s, t; edge_sd(es, ed, i, s, t);
    float ex = __expf(ebuf[i] - segmax[t]);
    ebuf[i] = ex;
    atomicAdd(&segsum[t], ex);
}

// agg[dst,:] += alpha * hp[src,:]   (wave per edge, lane strides the feature dim)
__global__ void k_edge_scatter(const int* __restrict__ es, const int* __restrict__ ed,
                               const float* __restrict__ ebuf, const float* __restrict__ segsum,
                               const float* __restrict__ hp, float* __restrict__ agg, int C) {
    int w    = (blockIdx.x * blockDim.x + threadIdx.x) >> 5;
    int lane = threadIdx.x & 31;
    if (w >= EN_) return;
    int s, t; edge_sd(es, ed, w, s, t);
    float a = ebuf[w] / (segsum[t] + EPS_);
    const float* hs = hp  + (size_t)s * C;
    float*       at = agg + (size_t)t * C;
    for (int c = lane; c < C; c += 32) atomicAdd(&at[c], a * hs[c]);
}

// layer-1 finish: h_bf16 = bf16(elu(agg + b))
__global__ void k_finish1(const float* __restrict__ agg, const float* __restrict__ b,
                          uint16_t* __restrict__ hbf, int C, size_t total) {
    size_t i = blockIdx.x * (size_t)blockDim.x + threadIdx.x;
    if (i >= total) return;
    float x = agg[i] + b[i % C];
    x = (x > 0.f) ? x : (__expf(x) - 1.0f);     // ELU(alpha=1)
    hbf[i] = f32_to_bf16(x);
}

// layer-2 finish: stacked[v] = agg + b
__global__ void k_finish2(const float* __restrict__ agg, const float* __restrict__ b,
                          float* __restrict__ stk, int C, size_t total) {
    size_t i = blockIdx.x * (size_t)blockDim.x + threadIdx.x;
    if (i >= total) return;
    stk[i] = agg[i] + b[i % C];
}

// ---------------- view attention + normalize (wave per node) ----------------
__global__ void k_view_attn(const float* __restrict__ stacked,   // [3,N,128]
                            const float* __restrict__ vw1,       // [128,32]
                            const float* __restrict__ vb1,       // [32]
                            const float* __restrict__ vw2,       // [32]
                            const float* __restrict__ vb2,       // [1]
                            float* __restrict__ z_out,           // [N,128]
                            float* __restrict__ vw_out,          // [N,3]
                            uint16_t* __restrict__ znbf,         // [N,128] bf16
                            int N) {
    int n    = (blockIdx.x * blockDim.x + threadIdx.x) >> 5;
    int lane = threadIdx.x & 31;
    if (n >= N) return;

    float sc[3];
    #pragma unroll
    for (int v = 0; v < 3; v++) {
        const float* row = stacked + ((size_t)v * N + n) * D_;
        float hj = vb1[lane];                       // hidden dim == 32 == wave width
        for (int d = 0; d < D_; d++) hj += row[d] * vw1[d * 32 + lane];
        hj = (hj > 0.f) ? hj : 0.f;
        float p = hj * vw2[lane];
        #pragma unroll
        for (int off = 16; off; off >>= 1) p += __shfl_down(p, off, 32);
        p += vb2[0];
        sc[v] = __shfl(p, 0, 32);
    }
    float m  = fmaxf(sc[0], fmaxf(sc[1], sc[2]));
    float w0 = __expf(sc[0] - m), w1 = __expf(sc[1] - m), w2 = __expf(sc[2] - m);
    float ws = w0 + w1 + w2;
    w0 /= ws; w1 /= ws; w2 /= ws;
    if (lane == 0) { vw_out[(size_t)n * 3 + 0] = w0; vw_out[(size_t)n * 3 + 1] = w1;
                     vw_out[(size_t)n * 3 + 2] = w2; }

    const float* s0 = stacked + ((size_t)0 * N + n) * D_;
    const float* s1 = stacked + ((size_t)1 * N + n) * D_;
    const float* s2 = stacked + ((size_t)2 * N + n) * D_;
    float zr[4]; float ss = 0.f;
    #pragma unroll
    for (int q = 0; q < 4; q++) {
        int d = lane + 32 * q;
        float z = w0 * s0[d] + w1 * s1[d] + w2 * s2[d];
        z_out[(size_t)n * D_ + d] = z;
        float r = (z > 0.f) ? z : 0.f;
        zr[q] = r; ss += r * r;
    }
    #pragma unroll
    for (int off = 16; off; off >>= 1) ss += __shfl_xor(ss, off, 32);
    float inv = 1.0f / fmaxf(sqrtf(ss), 1e-7f);
    #pragma unroll
    for (int q = 0; q < 4; q++)
        znbf[(size_t)n * D_ + lane + 32 * q] = f32_to_bf16(zr[q] * inv);
}

// ---------------- host side ----------------
extern "C" void kernel_launch(void* const* d_in, const int* in_sizes, int n_in,
                              void* d_out, int out_size, void* d_ws, size_t ws_size,
                              hipStream_t stream) {
    const float* x   = (const float*)d_in[0];
    const int*   ei  = (const int*)  d_in[1];    // [V,2,E]
    const float* W1  = (const float*)d_in[2];    // [V,256,256]
    const float* as1 = (const float*)d_in[3];
    const float* ad1 = (const float*)d_in[4];
    const float* b1  = (const float*)d_in[5];
    const float* W2  = (const float*)d_in[6];    // [V,256,128]
    const float* as2 = (const float*)d_in[7];
    const float* ad2 = (const float*)d_in[8];
    const float* b2  = (const float*)d_in[9];
    const float* vw1 = (const float*)d_in[10];
    const float* vb1 = (const float*)d_in[11];
    const float* vw2 = (const float*)d_in[12];
    const float* vb2 = (const float*)d_in[13];

    float* out    = (float*)d_out;
    float* z_out  = out;                                   // [N,128]
    float* recon  = out + (size_t)N_ * D_;                 // [N,N]
    float* vw_out = recon + (size_t)N_ * (size_t)N_;       // [N,3]

    // workspace carve-up
    char*  ws  = (char*)d_ws;
    size_t off = 0;
    auto alloc = [&](size_t bytes) -> void* {
        void* p = ws + off;
        off += (bytes + 255) & ~(size_t)255;
        return p;
    };
    uint16_t* x_bf   = (uint16_t*)alloc((size_t)N_ * FIN_ * 2);
    uint16_t* wt_bf  = (uint16_t*)alloc((size_t)256 * 256 * 2);
    float*    hp     = (float*)   alloc((size_t)N_ * H_ * 4);
    float*    e_s    = (float*)   alloc((size_t)N_ * 4);
    float*    e_d    = (float*)   alloc((size_t)N_ * 4);
    float*    smax   = (float*)   alloc((size_t)N_ * 4);
    float*    ssum   = (float*)   alloc((size_t)N_ * 4);
    float*    ebuf   = (float*)   alloc((size_t)EN_ * 4);
    float*    agg    = (float*)   alloc((size_t)N_ * H_ * 4);
    uint16_t* h_bf   = (uint16_t*)alloc((size_t)N_ * H_ * 2);
    float*    stk    = (float*)   alloc((size_t)Vv_ * N_ * D_ * 4);
    uint16_t* znbf   = (uint16_t*)alloc((size_t)N_ * D_ * 2);
    (void)ws_size; (void)in_sizes; (void)n_in; (void)out_size;

    const int T = 256;
    auto blks = [](size_t n, int t) { return (unsigned)((n + t - 1) / t); };

    // x -> bf16 once
    k_cvt_bf16<<<blks((size_t)N_ * FIN_, T), T, 0, stream>>>(x, x_bf, (size_t)N_ * FIN_);

    for (int v = 0; v < Vv_; v++) {
        const int* src = ei + ((size_t)v * 2 + 0) * E_;
        const int* dst = ei + ((size_t)v * 2 + 1) * E_;

        // ---------- GAT layer 1 (C = 256) ----------
        k_transpose_cvt<<<blks(256 * 256, T), T, 0, stream>>>(W1 + (size_t)v * 256 * 256, wt_bf, 256, 256);
        {
            dim3 g(H_ / 128, N_ / 128);
            k_gemm_nt_bf16<<<g, T, 0, stream>>>(x_bf, wt_bf, hp, N_, H_, FIN_, 0);
        }
        k_node_dots<<<N_ / 8, T, 0, stream>>>(hp, as1 + (size_t)v * H_, ad1 + (size_t)v * H_,
                                              e_s, e_d, N_, H_);
        k_fill_f32<<<blks(N_, T), T, 0, stream>>>(smax, -3.0e38f, N_);
        k_fill_f32<<<blks(N_, T), T, 0, stream>>>(ssum, 0.f, N_);
        k_fill_f32<<<blks((size_t)N_ * H_, T), T, 0, stream>>>(agg, 0.f, (size_t)N_ * H_);
        k_edge_max   <<<blks(EN_, T), T, 0, stream>>>(src, dst, e_s, e_d, ebuf, smax);
        k_edge_expsum<<<blks(EN_, T), T, 0, stream>>>(src, dst, ebuf, smax, ssum);
        k_edge_scatter<<<EN_ / 8, T, 0, stream>>>(src, dst, ebuf, ssum, hp, agg, H_);
        k_finish1<<<blks((size_t)N_ * H_, T), T, 0, stream>>>(agg, b1 + (size_t)v * H_, h_bf,
                                                              H_, (size_t)N_ * H_);

        // ---------- GAT layer 2 (C = 128) ----------
        k_transpose_cvt<<<blks(256 * 128, T), T, 0, stream>>>(W2 + (size_t)v * 256 * 128, wt_bf, 256, 128);
        {
            dim3 g(D_ / 128, N_ / 128);
            k_gemm_nt_bf16<<<g, T, 0, stream>>>(h_bf, wt_bf, hp, N_, D_, H_, 0);
        }
        k_node_dots<<<N_ / 8, T, 0, stream>>>(hp, as2 + (size_t)v * D_, ad2 + (size_t)v * D_,
                                              e_s, e_d, N_, D_);
        k_fill_f32<<<blks(N_, T), T, 0, stream>>>(smax, -3.0e38f, N_);
        k_fill_f32<<<blks(N_, T), T, 0, stream>>>(ssum, 0.f, N_);
        k_fill_f32<<<blks((size_t)N_ * D_, T), T, 0, stream>>>(agg, 0.f, (size_t)N_ * D_);
        k_edge_max   <<<blks(EN_, T), T, 0, stream>>>(src, dst, e_s, e_d, ebuf, smax);
        k_edge_expsum<<<blks(EN_, T), T, 0, stream>>>(src, dst, ebuf, smax, ssum);
        k_edge_scatter<<<EN_ / 8, T, 0, stream>>>(src, dst, ebuf, ssum, hp, agg, D_);
        k_finish2<<<blks((size_t)N_ * D_, T), T, 0, stream>>>(agg, b2 + (size_t)v * D_,
                                                              stk + (size_t)v * N_ * D_,
                                                              D_, (size_t)N_ * D_);
    }

    // ---------- view attention, fuse, normalize ----------
    k_view_attn<<<N_ / 8, T, 0, stream>>>(stk, vw1, vb1, vw2, vb2, z_out, vw_out, znbf, N_);

    // ---------- recon = sigmoid(zn @ zn^T), [12288 x 12288], K=128 ----------
    {
        dim3 g(N_ / 128, N_ / 128);
        k_gemm_nt_bf16<<<g, T, 0, stream>>>(znbf, znbf, recon, N_, N_, D_, 1);
    }
}
